// CGNet_SSM_36172214567198
// MI455X (gfx1250) — compile-verified, matrix operand order checked by reading, not run on hardware
//
#include <hip/hip_runtime.h>
#include <hip/hip_bf16.h>

typedef _Float16 f16;
typedef __attribute__((ext_vector_type(16))) _Float16 v16h;
typedef __attribute__((ext_vector_type(8)))  _Float16 h8v;
typedef __attribute__((ext_vector_type(8)))  float    v8f;
typedef int v4i __attribute__((vector_size(16)));

// Async global->LDS (CDNA5): gate on builtin availability so the file compiles
// on any toolchain; fallback is plain guarded LDS stores.
#if defined(__gfx1250__) && __has_builtin(__builtin_amdgcn_global_load_async_to_lds_b128) && __has_builtin(__builtin_amdgcn_s_wait_asynccnt)
#define CG_HAS_ASYNC 1
typedef __attribute__((address_space(1))) v4i* cg_gp_t;
typedef __attribute__((address_space(3))) v4i* cg_lp_t;
#else
#define CG_HAS_ASYNC 0
#endif

// ---------------------------------------------------------------------------
// Implicit-GEMM convolution, D(f32) = A(f16) x B(f16) via WMMA 16x16x32.
// M = Nb*H*W output pixels, N = Co, K = Ci*KH*KW.
// Block: 128 threads (4 waves). Tile 64(M) x 64(N); per wave: 16 N-cols,
// 4 M-subtiles -> 4 WMMAs per K-step of 32.
// H, W passed as log2 (all spatial sizes are powers of two here);
// KH/KW/PAD are template constants (3,3,1) or (1,1,0).
// ---------------------------------------------------------------------------
template<int KH, int KW, int PAD>
__global__ void __launch_bounds__(128)
cg_conv_wmma(const f16* __restrict__ x, const f16* __restrict__ wgt,
             const float* __restrict__ scale, const float* __restrict__ bias,
             f16* __restrict__ y16,
             int Nb, int Ci, int logH, int logW, int Co, int relu)
{
    const int H = 1 << logH, W = 1 << logW;
    const int M = Nb << (logH + logW);
    constexpr int KR = KH * KW;
    const int K = Ci * KR;
    const int m0 = blockIdx.x * 64;
    const int n0 = blockIdx.y * 64;
    const int tid  = threadIdx.x;
    const int wave = tid >> 5;
    const int lane = tid & 31;
    const int half = lane >> 4;
    const int r    = lane & 15;

    __shared__ __align__(16) f16 As[64 * 32];   // [m][k]
    __shared__ __align__(16) f16 Bs[64 * 32];   // [n][k] (weights, n-major)

    // full B tile without bounds checks? (true for all layers but stem/head)
    const bool fastB = ((K & 31) == 0) && (n0 + 64 <= Co);

    v8f acc0 = {}, acc1 = {}, acc2 = {}, acc3 = {};

    for (int k0 = 0; k0 < K; k0 += 32) {
        // ---- stage A tile (im2col on the fly) ----
        for (int i = tid; i < 64 * 32; i += 128) {
            int mi = i >> 5, kk = i & 31;
            int k  = k0 + kk;
            int m  = m0 + mi;
            f16 v = (f16)0;
            if (k < K && m < M) {
                int wo = m & (W - 1);
                int ho = (m >> logW) & (H - 1);
                int nb = m >> (logW + logH);
                int ci, dy, dx;
                if (KR == 1) { ci = k; dy = 0; dx = 0; }
                else {
                    ci = k / KR;                 // constexpr divisor -> mul/shift
                    int kr = k - ci * KR;
                    dy = kr / KW; dx = kr - dy * KW;
                }
                int hy = ho + dy - PAD, wx = wo + dx - PAD;
                if (hy >= 0 && hy < H && wx >= 0 && wx < W)
                    v = x[(((size_t)nb * Ci + ci) << (logH + logW)) + (hy << logW) + wx];
            }
            As[i] = v;
        }
        // ---- stage B tile (f16 weights, contiguous 64B per row) ----
        if (fastB) {
#if CG_HAS_ASYNC
            // 64 rows x 64B = 256 chunks of 16B; 2 per thread, async to LDS.
            #pragma unroll
            for (int c = 0; c < 2; ++c) {
                int idx = tid + c * 128;
                int ni = idx >> 2;
                int ck = idx & 3;
                const f16* g = wgt + (size_t)(n0 + ni) * K + k0 + ck * 8;
                f16* l = &Bs[ni * 32 + ck * 8];
                __builtin_amdgcn_global_load_async_to_lds_b128(
                    (cg_gp_t)g, (cg_lp_t)l, 0, 0);
            }
            __builtin_amdgcn_s_wait_asynccnt(0);
#else
            #pragma unroll
            for (int c = 0; c < 2; ++c) {
                int idx = tid + c * 128;
                int ni = idx >> 2;
                int ck = idx & 3;
                *(h8v*)&Bs[ni * 32 + ck * 8] =
                    *(const h8v*)(wgt + (size_t)(n0 + ni) * K + k0 + ck * 8);
            }
#endif
        } else {
            for (int i = tid; i < 64 * 32; i += 128) {
                int ni = i >> 5, kk = i & 31;
                int k  = k0 + kk;
                int n  = n0 + ni;
                f16 v = (f16)0;
                if (k < K && n < Co) {
                    v = wgt[(size_t)n * K + k];
                    if (k + 32 < K)
                        __builtin_prefetch(&wgt[(size_t)n * K + k + 32], 0, 1);
                }
                Bs[i] = v;
            }
        }
        __syncthreads();

        // ---- B fragment: N = wave*16 + r, K split by lane half ----
        h8v blo = *(const h8v*)&Bs[(wave * 16 + r) * 32 + half * 8];
        h8v bhi = *(const h8v*)&Bs[(wave * 16 + r) * 32 + 16 + half * 8];
        v16h bf;
#pragma unroll
        for (int i = 0; i < 8; ++i) { bf[i] = blo[i]; bf[i + 8] = bhi[i]; }

#pragma unroll
        for (int s = 0; s < 4; ++s) {
            h8v alo = *(const h8v*)&As[(s * 16 + r) * 32 + half * 8];
            h8v ahi = *(const h8v*)&As[(s * 16 + r) * 32 + 16 + half * 8];
            v16h af;
#pragma unroll
            for (int i = 0; i < 8; ++i) { af[i] = alo[i]; af[i + 8] = ahi[i]; }
            v8f& acc = (s == 0) ? acc0 : (s == 1) ? acc1 : (s == 2) ? acc2 : acc3;
            acc = __builtin_amdgcn_wmma_f32_16x16x32_f16(
                false, af, false, bf, (short)0, acc, false, false);
        }
        __syncthreads();
    }

    // ---- epilogue: D layout VGPR i -> M = i + 8*half, N = r ----
    int n = n0 + wave * 16 + r;
    float sc = 0.f, bi = 0.f;
    if (n < Co) { sc = scale[n]; bi = bias[n]; }
#pragma unroll
    for (int s = 0; s < 4; ++s) {
        v8f acc = (s == 0) ? acc0 : (s == 1) ? acc1 : (s == 2) ? acc2 : acc3;
#pragma unroll
        for (int i = 0; i < 8; ++i) {
            int m = m0 + s * 16 + i + 8 * half;
            if (m < M && n < Co) {
                float v = acc[i] * sc + bi;
                if (relu) v = fmaxf(v, 0.f);
                int wo = m & (W - 1);
                int ho = (m >> logW) & (H - 1);
                int nb = m >> (logW + logH);
                y16[(((size_t)nb * Co + n) << (logH + logW)) + (ho << logW) + wo] = (f16)v;
            }
        }
    }
}

// ---------------------------------------------------------------------------
// Small helper kernels
// ---------------------------------------------------------------------------
__global__ void cg_fold_bn(const float* g, const float* b, const float* m,
                           const float* v, float* sc, float* bi, int C)
{
    int i = blockIdx.x * blockDim.x + threadIdx.x;
    if (i < C) {
        float s = g[i] * rsqrtf(v[i] + 1e-5f);
        sc[i] = s;
        bi[i] = b[i] - m[i] * s;
    }
}

__global__ void cg_set_bias(const float* b, float* sc, float* bi, int C)
{
    int i = blockIdx.x * blockDim.x + threadIdx.x;
    if (i < C) { sc[i] = 1.f; bi[i] = b[i]; }
}

__global__ void cg_cvt_f32_f16(const float* x, f16* y, int n)
{
    int i = blockIdx.x * blockDim.x + threadIdx.x;
    if (i < n) y[i] = (f16)x[i];
}

__global__ void cg_relu_pool(const f16* x, f16* y, int NC, int H, int W)
{
    int Ho = H >> 1, Wo = W >> 1;
    int idx = blockIdx.x * blockDim.x + threadIdx.x;
    int total = NC * Ho * Wo;
    if (idx >= total) return;
    int wo = idx % Wo; int ho = (idx / Wo) % Ho; int p = idx / (Wo * Ho);
    const f16* xp = x + (size_t)p * H * W + (size_t)(2 * ho) * W + 2 * wo;
    float a = (float)xp[0], b = (float)xp[1];
    float c = (float)xp[W], d = (float)xp[W + 1];
    float m = fmaxf(fmaxf(a, b), fmaxf(c, d));
    y[idx] = (f16)fmaxf(m, 0.f);
}

__global__ void cg_concat2(const f16* a, const f16* b, f16* y,
                           int N, int Ca, int Cb, int HW)
{
    int idx = blockIdx.x * blockDim.x + threadIdx.x;
    int Cs = Ca + Cb;
    int total = N * Cs * HW;
    if (idx >= total) return;
    int hw = idx % HW; int c = (idx / HW) % Cs; int n = idx / (HW * Cs);
    y[idx] = (c < Ca) ? a[((size_t)n * Ca + c) * HW + hw]
                      : b[((size_t)n * Cb + (c - Ca)) * HW + hw];
}

// bilinear, align_corners=True
__global__ void cg_resize_h(const f16* x, f16* y, int NC, int IH, int IW,
                            int OH, int OW)
{
    int idx = blockIdx.x * blockDim.x + threadIdx.x;
    int total = NC * OH * OW;
    if (idx >= total) return;
    int ow = idx % OW; int oh = (idx / OW) % OH; int p = idx / (OW * OH);
    float fy = (OH > 1) ? (float)oh * (float)(IH - 1) / (float)(OH - 1) : 0.f;
    float fx = (OW > 1) ? (float)ow * (float)(IW - 1) / (float)(OW - 1) : 0.f;
    int y0 = (int)floorf(fy), x0 = (int)floorf(fx);
    int y1 = (y0 + 1 < IH) ? y0 + 1 : IH - 1;
    int x1 = (x0 + 1 < IW) ? x0 + 1 : IW - 1;
    float wy = fy - (float)y0, wx = fx - (float)x0;
    const f16* xp = x + (size_t)p * IH * IW;
    float v00 = (float)xp[y0 * IW + x0], v01 = (float)xp[y0 * IW + x1];
    float v10 = (float)xp[y1 * IW + x0], v11 = (float)xp[y1 * IW + x1];
    float v = (v00 * (1.f - wx) + v01 * wx) * (1.f - wy)
            + (v10 * (1.f - wx) + v11 * wx) * wy;
    y[idx] = (f16)v;
}

__global__ void cg_resize_f(const f16* x, float* y, int NC, int IH, int IW,
                            int OH, int OW)
{
    int idx = blockIdx.x * blockDim.x + threadIdx.x;
    int total = NC * OH * OW;
    if (idx >= total) return;
    int ow = idx % OW; int oh = (idx / OW) % OH; int p = idx / (OW * OH);
    float fy = (OH > 1) ? (float)oh * (float)(IH - 1) / (float)(OH - 1) : 0.f;
    float fx = (OW > 1) ? (float)ow * (float)(IW - 1) / (float)(OW - 1) : 0.f;
    int y0 = (int)floorf(fy), x0 = (int)floorf(fx);
    int y1 = (y0 + 1 < IH) ? y0 + 1 : IH - 1;
    int x1 = (x0 + 1 < IW) ? x0 + 1 : IW - 1;
    float wy = fy - (float)y0, wx = fx - (float)x0;
    const f16* xp = x + (size_t)p * IH * IW;
    float v00 = (float)xp[y0 * IW + x0], v01 = (float)xp[y0 * IW + x1];
    float v10 = (float)xp[y1 * IW + x0], v11 = (float)xp[y1 * IW + x1];
    y[idx] = (v00 * (1.f - wx) + v01 * wx) * (1.f - wy)
           + (v10 * (1.f - wx) + v11 * wx) * wy;
}

// F_mod = F + alpha * prior (prior broadcast over channels)
__global__ void cg_add_prior(const f16* F, const f16* prior, f16* out,
                             int N, int C, int HW, const float* alpha)
{
    int idx = blockIdx.x * blockDim.x + threadIdx.x;
    int total = N * C * HW;
    if (idx >= total) return;
    int hw = idx % HW; int n = idx / (HW * C);
    float a = alpha[0];
    out[idx] = (f16)((float)F[idx] + a * (float)prior[(size_t)n * HW + hw]);
}

// y = a + gamma * b
__global__ void cg_add_scaled(const f16* a, const f16* b, f16* y, int n,
                              const float* gamma)
{
    int i = blockIdx.x * blockDim.x + threadIdx.x;
    if (i < n) y[i] = (f16)((float)a[i] + gamma[0] * (float)b[i]);
}

// Directional linear scans: h = tanh(A[c])*h + B[c]*x, all states written.
// dir 0: l->r, 1: r->l (over W); 2: t->b, 3: b->t (over H).
// Output into channel block dir*hd of the (N, 4*hd, H, W) tensor.
__global__ void cg_ssm_scan(const f16* __restrict__ x, const float* __restrict__ Av,
                            const float* __restrict__ Bv, f16* __restrict__ hcat,
                            int N, int hd, int H, int W, int dir)
{
    int idx = blockIdx.x * blockDim.x + threadIdx.x;
    int L = (dir < 2) ? H : W;
    int T = (dir < 2) ? W : H;
    int lines = N * hd * L;
    if (idx >= lines) return;
    int l = idx % L; int c = (idx / L) % hd; int n = idx / (L * hd);
    float a = tanhf(Av[c]);
    float b = Bv[c];
    size_t ibase, obase; int stride;
    size_t oc = (size_t)n * 4 * hd + (size_t)dir * hd + c;
    if (dir < 2) { ibase = ((size_t)(n * hd + c) * H + l) * W; obase = (oc * H + l) * W; stride = 1; }
    else         { ibase = (size_t)(n * hd + c) * H * W + l;   obase = oc * H * W + l;   stride = W; }
    float h = 0.f;
    for (int t = 0; t < T; ++t) {
        int p = (dir == 1 || dir == 3) ? (T - 1 - t) : t;
        h = a * h + b * (float)x[ibase + (size_t)p * stride];
        hcat[obase + (size_t)p * stride] = (f16)h;
    }
}

// ---------------------------------------------------------------------------
// Host orchestration
// ---------------------------------------------------------------------------
extern "C" void kernel_launch(void* const* d_in, const int* in_sizes, int n_in,
                              void* d_out, int out_size, void* d_ws, size_t ws_size,
                              hipStream_t stream)
{
    (void)in_sizes; (void)n_in; (void)out_size; (void)ws_size;
    const int NB = 4;

    auto F = [&](int i) -> const float* { return (const float*)d_in[i]; };
    // d_in layout (depth-first dict flattening):
    // 0:A 1:B | vgg conv i: w=2+5i, bn g/b/m/v follow | reduce1..4 @67,72,77,82
    // dec1@87 dec2 w=92 b=93 | df1@94 df2 w=99 b=100
    // ssm1@101 ssm2@111 ssm3@121 (pin.w,pin.b,A,B,alpha,gamma,pout.w,pout.b,fus.w,fus.b)
    // dm4@131 dm3@136 dm2@141
    auto vggW = [](int i) { return 2 + 5 * i; };

    struct Bump {
        char* p; size_t off;
        void* take(size_t bytes) {
            void* r = p + off;
            off = (off + bytes + 255) & ~(size_t)255;
            return r;
        }
    } ws{(char*)d_ws, 0};
    auto takeH = [&](size_t n) { return (f16*)ws.take(n * sizeof(f16)); };
    auto takeF = [&](size_t n) { return (float*)ws.take(n * sizeof(float)); };

    float* sc = takeF(1024);
    float* bi = takeF(1024);
    f16* wbuf = takeH((size_t)512 * 1024 * 9);        // largest weight (dm4/reduce3/4)

    f16* xin = takeH((size_t)NB * 3 * 256 * 256);
    f16* P   = takeH((size_t)NB * 384 * 128 * 128);   // largest concat
    f16* Q   = takeH((size_t)NB * 64 * 256 * 256);
    f16* l1A = takeH((size_t)NB * 128 * 128 * 128);
    f16* l2A = takeH((size_t)NB * 256 * 64 * 64);
    f16* l3A = takeH((size_t)NB * 512 * 32 * 32);
    f16* l4A = takeH((size_t)NB * 512 * 16 * 16);
    f16* l1B = takeH((size_t)NB * 128 * 128 * 128);
    f16* l2B = takeH((size_t)NB * 256 * 64 * 64);
    f16* l3B = takeH((size_t)NB * 512 * 32 * 32);
    f16* l4B = takeH((size_t)NB * 512 * 16 * 16);
    f16* r1  = takeH((size_t)NB * 128 * 128 * 128);
    f16* r2  = takeH((size_t)NB * 256 * 64 * 64);
    f16* r3  = takeH((size_t)NB * 512 * 32 * 32);
    f16* r4  = takeH((size_t)NB * 512 * 16 * 16);
    f16* coarse = takeH((size_t)NB * 16 * 16);
    f16* prior  = takeH((size_t)NB * 128 * 128);
    f16* f4b = takeH((size_t)NB * 512 * 32 * 32);
    f16* f3b = takeH((size_t)NB * 256 * 64 * 64);
    f16* f2b = takeH((size_t)NB * 128 * 128 * 128);
    f16* fmod = takeH((size_t)NB * 128 * 128 * 128);
    f16* xh   = takeH((size_t)NB * 64 * 128 * 128);
    f16* hcat = takeH((size_t)NB * 256 * 128 * 128);
    f16* hf   = takeH((size_t)NB * 64 * 128 * 128);
    f16* pob  = takeH((size_t)NB * 128 * 128 * 128);
    f16* fm   = takeH((size_t)NB * 128 * 128);

    // conv launcher: converts weights to f16 then runs the WMMA conv.
    auto conv = [&](const f16* x, const float* w32, f16* y,
                    int Ci, int logH, int logW, int Co, int k3, int relu) {
        int wElems = Co * Ci * (k3 ? 9 : 1);
        cg_cvt_f32_f16<<<(wElems + 255) / 256, 256, 0, stream>>>(w32, wbuf, wElems);
        int M = NB << (logH + logW);
        dim3 grid((M + 63) / 64, (Co + 63) / 64);
        if (k3)
            cg_conv_wmma<3, 3, 1><<<grid, 128, 0, stream>>>(
                x, wbuf, sc, bi, y, NB, Ci, logH, logW, Co, relu);
        else
            cg_conv_wmma<1, 1, 0><<<grid, 128, 0, stream>>>(
                x, wbuf, sc, bi, y, NB, Ci, logH, logW, Co, relu);
    };
    auto conv_bn = [&](const f16* x, int wi, f16* y,
                       int Ci, int logH, int logW, int Co, int relu) {
        cg_fold_bn<<<(Co + 127) / 128, 128, 0, stream>>>(
            F(wi + 1), F(wi + 2), F(wi + 3), F(wi + 4), sc, bi, Co);
        conv(x, F(wi), y, Ci, logH, logW, Co, 1 /*3x3*/, relu);
    };
    auto conv_b = [&](const f16* x, int wi, int bidx, f16* y,
                      int Ci, int logH, int logW, int Co, int k3) {
        cg_set_bias<<<(Co + 127) / 128, 128, 0, stream>>>(F(bidx), sc, bi, Co);
        conv(x, F(wi), y, Ci, logH, logW, Co, k3, 0);
    };
    auto pool = [&](const f16* x, f16* y, int NC, int H, int W) {
        int total = NC * (H / 2) * (W / 2);
        cg_relu_pool<<<(total + 255) / 256, 256, 0, stream>>>(x, y, NC, H, W);
    };
    auto resize_h = [&](const f16* x, f16* y, int NC, int IH, int IW, int OH, int OW) {
        int total = NC * OH * OW;
        cg_resize_h<<<(total + 255) / 256, 256, 0, stream>>>(x, y, NC, IH, IW, OH, OW);
    };
    auto concat = [&](const f16* a, const f16* b, f16* y, int Ca, int Cb, int HW) {
        int total = NB * (Ca + Cb) * HW;
        cg_concat2<<<(total + 255) / 256, 256, 0, stream>>>(a, b, y, NB, Ca, Cb, HW);
    };

    auto backbone = [&](const float* img, f16* l1, f16* l2, f16* l3, f16* l4) {
        int n = NB * 3 * 256 * 256;
        cg_cvt_f32_f16<<<(n + 255) / 256, 256, 0, stream>>>(img, xin, n);
        conv_bn(xin, vggW(0), P, 3, 8, 8, 64, 1);
        conv_bn(P, vggW(1), Q, 64, 8, 8, 64, 0);
        pool(Q, P, NB * 64, 256, 256);
        conv_bn(P, vggW(2), Q, 64, 7, 7, 128, 1);
        conv_bn(Q, vggW(3), l1, 128, 7, 7, 128, 0);
        pool(l1, P, NB * 128, 128, 128);
        conv_bn(P, vggW(4), Q, 128, 6, 6, 256, 1);
        conv_bn(Q, vggW(5), P, 256, 6, 6, 256, 1);
        conv_bn(P, vggW(6), l2, 256, 6, 6, 256, 0);
        pool(l2, P, NB * 256, 64, 64);
        conv_bn(P, vggW(7), Q, 256, 5, 5, 512, 1);
        conv_bn(Q, vggW(8), P, 512, 5, 5, 512, 1);
        conv_bn(P, vggW(9), l3, 512, 5, 5, 512, 0);
        pool(l3, P, NB * 512, 32, 32);
        conv_bn(P, vggW(10), Q, 512, 4, 4, 512, 1);
        conv_bn(Q, vggW(11), P, 512, 4, 4, 512, 1);
        conv_bn(P, vggW(12), l4, 512, 4, 4, 512, 0);
    };

    auto run_ssm = [&](f16* feat, const f16* pr, int base,
                       int C, int hd, int logHW /*log2 of H==W*/) {
        int H = 1 << logHW, W = H, HW = H * W;
        int tot = NB * C * HW;
        cg_add_prior<<<(tot + 255) / 256, 256, 0, stream>>>(feat, pr, fmod, NB, C, HW, F(base + 4));
        cg_set_bias<<<(hd + 127) / 128, 128, 0, stream>>>(F(base + 1), sc, bi, hd);
        conv(fmod, F(base + 0), xh, C, logHW, logHW, hd, 0, 0);
        for (int dir = 0; dir < 4; ++dir) {
            int lines = NB * hd * H;
            cg_ssm_scan<<<(lines + 127) / 128, 128, 0, stream>>>(
                xh, F(base + 2), F(base + 3), hcat, NB, hd, H, W, dir);
        }
        cg_set_bias<<<(hd + 127) / 128, 128, 0, stream>>>(F(base + 9), sc, bi, hd);
        conv(hcat, F(base + 8), hf, 4 * hd, logHW, logHW, hd, 0, 0);
        cg_set_bias<<<(C + 127) / 128, 128, 0, stream>>>(F(base + 7), sc, bi, C);
        conv(hf, F(base + 6), pob, hd, logHW, logHW, C, 0, 0);
        cg_add_scaled<<<(tot + 255) / 256, 256, 0, stream>>>(fmod, pob, feat, tot, F(base + 5));
    };

    // ---- backbones ----
    backbone((const float*)d_in[0], l1A, l2A, l3A, l4A);
    backbone((const float*)d_in[1], l1B, l2B, l3B, l4B);

    // ---- reduce ----
    concat(l1A, l1B, P, 128, 128, 128 * 128);  conv_bn(P, 67, r1, 256, 7, 7, 128, 1);
    concat(l2A, l2B, P, 256, 256, 64 * 64);    conv_bn(P, 72, r2, 512, 6, 6, 256, 1);
    concat(l3A, l3B, P, 512, 512, 32 * 32);    conv_bn(P, 77, r3, 1024, 5, 5, 512, 1);
    concat(l4A, l4B, P, 512, 512, 16 * 16);    conv_bn(P, 82, r4, 1024, 4, 4, 512, 1);

    // ---- coarse decoder head ----
    conv_bn(r4, 87, Q, 512, 4, 4, 64, 1);
    conv_b(Q, 92, 93, coarse, 64, 4, 4, 1, 1 /*3x3*/);

    // ---- level 3 ----
    resize_h(coarse, prior, NB, 16, 16, 32, 32);
    run_ssm(r3, prior, 121, 512, 256, 5);
    resize_h(r4, Q, NB * 512, 16, 16, 32, 32);
    concat(Q, r3, P, 512, 512, 32 * 32);
    conv_bn(P, 131, f4b, 1024, 5, 5, 512, 1);

    // ---- level 2 ----
    resize_h(coarse, prior, NB, 16, 16, 64, 64);
    run_ssm(r2, prior, 111, 256, 128, 6);
    resize_h(f4b, Q, NB * 512, 32, 32, 64, 64);
    concat(Q, r2, P, 512, 256, 64 * 64);
    conv_bn(P, 136, f3b, 768, 6, 6, 256, 1);

    // ---- level 1 ----
    resize_h(coarse, prior, NB, 16, 16, 128, 128);
    run_ssm(r1, prior, 101, 128, 64, 7);
    resize_h(f3b, Q, NB * 256, 64, 64, 128, 128);
    concat(Q, r1, P, 256, 128, 128 * 128);
    conv_bn(P, 141, f2b, 384, 7, 7, 128, 1);

    // ---- outputs ----
    float* out = (float*)d_out;
    {
        int total = NB * 256 * 256;
        cg_resize_f<<<(total + 255) / 256, 256, 0, stream>>>(coarse, out, NB, 16, 16, 256, 256);
    }
    conv_bn(f2b, 94, Q, 128, 7, 7, 64, 1);
    conv_b(Q, 99, 100, fm, 64, 7, 7, 1, 0 /*1x1*/);
    {
        int total = NB * 256 * 256;
        cg_resize_f<<<(total + 255) / 256, 256, 0, stream>>>(fm, out + (size_t)NB * 256 * 256,
                                                             NB, 128, 128, 256, 256);
    }
}
